// DecoderLayer_26070451486896
// MI455X (gfx1250) — matrix operand order repaired and checked by reading, not compiled, not run on hardware
//
#include <hip/hip_runtime.h>
#include <hip/hip_bf16.h>
#include <math.h>

// ---------------------------------------------------------------------------
// Types for gfx1250 WMMA (wave32): v_wmma_f32_16x16x32_bf16
// ---------------------------------------------------------------------------
typedef __bf16 bf16;
typedef __attribute__((ext_vector_type(16))) __bf16 v16bf;
typedef __attribute__((ext_vector_type(8)))  __bf16 v8bf;
typedef __attribute__((ext_vector_type(8)))  float  v8f;

__device__ __forceinline__ v8f wmma_bf16(v16bf a, v16bf b, v8f c) {
  // 8 args: (neg_a, A, neg_b, B, c_mod, C, reuse_a, reuse_b)
  return __builtin_amdgcn_wmma_f32_16x16x32_bf16(false, a, false, b, (short)0, c,
                                                 false, false);
}

__device__ __forceinline__ v8f zero8() {
  v8f z = {0.f, 0.f, 0.f, 0.f, 0.f, 0.f, 0.f, 0.f};
  return z;
}

// ---------------------------------------------------------------------------
// Fragment loaders from LDS, matching the CDNA5 ISA VGPR layouts (CDNA5 ISA):
//  A (16x32 bf16): lane L -> row L%16; VGPR0-3 hold K = (L/16)*8 + 0..7,
//                  VGPR4-7 hold K = (L/16)*8 + 16..23.
//  B (32x16 bf16): lane L -> col L%16; same K layout as A (VGPR0-3 K = (L/16)*8 + 0..7, VGPR4-7 K = (L/16)*8 + 16..23).
// `stride` = LDS row stride in elements (multiple of 8 -> 16B-aligned b128s).
// ---------------------------------------------------------------------------
__device__ __forceinline__ v16bf frag_a(const bf16* base, int stride) {
  int lane = threadIdx.x & 31;
  const bf16* p = base + (lane & 15) * stride + ((lane >> 4) << 3);
  v8bf lo = *(const v8bf*)(p);
  v8bf hi = *(const v8bf*)(p + 16);
  return __builtin_shufflevector(lo, hi, 0, 1, 2, 3, 4, 5, 6, 7,
                                 8, 9, 10, 11, 12, 13, 14, 15);
}

__device__ __forceinline__ v16bf frag_b(const bf16* base, int stride) {
  int lane = threadIdx.x & 31;
  const bf16* p = base + (lane & 15) * stride + ((lane >> 4) << 3);
  v8bf lo = *(const v8bf*)(p);
  v8bf hi = *(const v8bf*)(p + 16);
  return __builtin_shufflevector(lo, hi, 0, 1, 2, 3, 4, 5, 6, 7,
                                 8, 9, 10, 11, 12, 13, 14, 15);
}

__device__ __forceinline__ void cvt_store4(bf16* d, float4 v) {
  d[0] = (bf16)v.x; d[1] = (bf16)v.y; d[2] = (bf16)v.z; d[3] = (bf16)v.w;
}

// ---------------------------------------------------------------------------
// GEMM (NT): C[M,N] = A[M,K] * W[N,K]^T (+bias[N]) (+ReLU).  f32 in/out, bf16
// WMMA compute.  Block = 256 thr = 8 waves (4x2), block tile 128x128, wave
// tile 32x64 (8 accumulators -> 1.5 ds_load_b128 per WMMA), K-step 32.
// Double-buffered LDS with register prefetch: global loads for tile t+1 are
// issued before the WMMA block for tile t; ONE barrier per K-iteration.
// All dims are exact multiples of the tiles (M in {2048,4096}, N,K in
// {512,1024,2048}).
// ---------------------------------------------------------------------------
#define GBM 128
#define GBN 128
#define GBK 32
#define GSTR (GBK + 8)  // 40 elements -> 80B row stride, 16B aligned

__global__ __launch_bounds__(256) void gemm_nt_kernel(
    const float* __restrict__ A, const float* __restrict__ W,
    const float* __restrict__ bias, float* __restrict__ C,
    int M, int N, int K, int relu) {
  __shared__ bf16 sA[2][GBM * GSTR];
  __shared__ bf16 sB[2][GBN * GSTR];

  const int tid  = threadIdx.x;
  const int m0   = blockIdx.y * GBM;
  const int n0   = blockIdx.x * GBN;
  const int wave = tid >> 5;
  const int lane = tid & 31;
  const int wm   = wave >> 1;  // 0..3 -> rows wm*32
  const int wn   = wave & 1;   // 0..1 -> cols wn*64

  // Staging map: each thread owns 16 f32 of A and 16 f32 of B per tile.
  const int sr = tid >> 1;        // 0..127
  const int sc = (tid & 1) * 16;  // 0 or 16
  const float* apg = A + (size_t)(m0 + sr) * K + sc;
  const float* bpg = W + (size_t)(n0 + sr) * K + sc;

  v8f acc[2][4];
#pragma unroll
  for (int i = 0; i < 2; ++i)
#pragma unroll
    for (int j = 0; j < 4; ++j) acc[i][j] = zero8();

  const int nk = K / GBK;

  // Stage tile 0 into buffer 0.
  {
    bf16* da = &sA[0][sr * GSTR + sc];
    bf16* db = &sB[0][sr * GSTR + sc];
#pragma unroll
    for (int q = 0; q < 4; ++q) {
      cvt_store4(da + q * 4, *(const float4*)(apg + q * 4));
      cvt_store4(db + q * 4, *(const float4*)(bpg + q * 4));
    }
  }
  __syncthreads();

  for (int t = 0; t < nk; ++t) {
    const int cur = t & 1;
    const bool pf = (t + 1 < nk);
    float4 pa[4], pb[4];
    if (pf) {  // issue next-tile global loads before the matrix block
      const float* ap = apg + (size_t)(t + 1) * GBK;
      const float* bp = bpg + (size_t)(t + 1) * GBK;
#pragma unroll
      for (int q = 0; q < 4; ++q) {
        pa[q] = *(const float4*)(ap + q * 4);
        pb[q] = *(const float4*)(bp + q * 4);
      }
    }

    const v16bf af0 = frag_a(&sA[cur][(wm * 32) * GSTR], GSTR);
    const v16bf af1 = frag_a(&sA[cur][(wm * 32 + 16) * GSTR], GSTR);
    v16bf bfr[4];
#pragma unroll
    for (int j = 0; j < 4; ++j)
      bfr[j] = frag_b(&sB[cur][(wn * 64 + j * 16) * GSTR], GSTR);
#pragma unroll
    for (int j = 0; j < 4; ++j) {
      acc[0][j] = wmma_bf16(af0, bfr[j], acc[0][j]);
      acc[1][j] = wmma_bf16(af1, bfr[j], acc[1][j]);
    }

    if (pf) {  // convert + store into the other buffer
      bf16* da = &sA[cur ^ 1][sr * GSTR + sc];
      bf16* db = &sB[cur ^ 1][sr * GSTR + sc];
#pragma unroll
      for (int q = 0; q < 4; ++q) {
        cvt_store4(da + q * 4, pa[q]);
        cvt_store4(db + q * 4, pb[q]);
      }
    }
    __syncthreads();
  }

  // Epilogue.  C layout: VGPR r -> row +r + 8*(lane/16), col +lane%16.
  const int ln = lane & 15, lg = lane >> 4;
  const int crow = m0 + wm * 32;
  const int ccol = n0 + wn * 64;
  float bv[4];
#pragma unroll
  for (int j = 0; j < 4; ++j) bv[j] = bias ? bias[ccol + j * 16 + ln] : 0.f;
#pragma unroll
  for (int i = 0; i < 2; ++i)
#pragma unroll
    for (int r = 0; r < 8; ++r) {
      const int row = crow + i * 16 + r + 8 * lg;
#pragma unroll
      for (int j = 0; j < 4; ++j) {
        float v = acc[i][j][r] + bv[j];
        if (relu) v = fmaxf(v, 0.f);
        C[(size_t)row * N + ccol + j * 16 + ln] = v;
      }
    }
}

// ---------------------------------------------------------------------------
// Fused flash attention (head_dim = 64).  Tensors laid out (B, L, H, 64).
// Block = 128 thr = 4 waves; block owns 64 query rows (16 per wave).
// K staged row-major, V staged transposed, both bf16 in double-buffered LDS
// with register prefetch -> ONE barrier per key block.  S = Q*K^T by WMMA,
// online softmax in f32 (16-lane xor butterflies match the C-matrix layout),
// P relaid out C->A through wave-private LDS (same-wave DS ops are in-order
// per the ISA, so no workgroup barrier is needed), O += P*V by WMMA.
// Q is pre-scaled by `scale`; causal mask is analytic (tril).
// ---------------------------------------------------------------------------
#define FSTR 72  // 144B row stride, 16B aligned

__global__ __launch_bounds__(128) void flash_attn_kernel(
    const float* __restrict__ Qb, const float* __restrict__ Kb,
    const float* __restrict__ Vb, float* __restrict__ Ob,
    int H, int LQ, int LK, int causal, float scale) {
  __shared__ bf16 sQ[64 * FSTR];
  __shared__ bf16 sK[2][64 * FSTR];
  __shared__ bf16 sVt[2][64 * FSTR];
  __shared__ bf16 sP[4 * 16 * FSTR];

  const int tid  = threadIdx.x;
  const int wave = tid >> 5;
  const int lane = tid & 31;
  const int ln   = lane & 15, lg = lane >> 4;
  const int qb   = blockIdx.x;
  const int z    = blockIdx.y;
  const int b    = z / H, h = z % H;
  const size_t ld = (size_t)H * 64;

  const float* Qg = Qb + ((size_t)b * LQ * H + h) * 64 + (size_t)(qb * 64) * ld;
  const float* Kg = Kb + ((size_t)b * LK * H + h) * 64;
  const float* Vg = Vb + ((size_t)b * LK * H + h) * 64;
  float*       Og = Ob + ((size_t)b * LQ * H + h) * 64 + (size_t)(qb * 64) * ld;

  // Per-thread staging map: 8 float4 of K + 8 float4 of V per 64x64 tile.
  // iteration u: linear float4 index i = tid + u*128 -> row i>>4, col (i&15)*4

  // Stage scaled Q and KV tile 0 (buffer 0).
#pragma unroll
  for (int u = 0; u < 8; ++u) {
    const int i = tid + u * 128, row = i >> 4, col = (i & 15) * 4;
    float4 q4 = *(const float4*)(Qg + (size_t)row * ld + col);
    bf16* d = &sQ[row * FSTR + col];
    d[0] = (bf16)(q4.x * scale); d[1] = (bf16)(q4.y * scale);
    d[2] = (bf16)(q4.z * scale); d[3] = (bf16)(q4.w * scale);
    const size_t goff = (size_t)row * ld + col;
    cvt_store4(&sK[0][row * FSTR + col], *(const float4*)(Kg + goff));
    float4 v4 = *(const float4*)(Vg + goff);
    sVt[0][(col + 0) * FSTR + row] = (bf16)v4.x;
    sVt[0][(col + 1) * FSTR + row] = (bf16)v4.y;
    sVt[0][(col + 2) * FSTR + row] = (bf16)v4.z;
    sVt[0][(col + 3) * FSTR + row] = (bf16)v4.w;
  }
  __syncthreads();
  const v16bf aq0 = frag_a(&sQ[(wave * 16) * FSTR], FSTR);       // d = 0..31
  const v16bf aq1 = frag_a(&sQ[(wave * 16) * FSTR + 32], FSTR);  // d = 32..63

  float mr[8], lr[8];
#pragma unroll
  for (int r = 0; r < 8; ++r) { mr[r] = -3.0e38f; lr[r] = 0.f; }
  v8f o0 = zero8(), o1 = zero8(), o2 = zero8(), o3 = zero8();

  const int nkb = causal ? (qb + 1) : (LK >> 6);
  for (int kb = 0; kb < nkb; ++kb) {
    const int cur = kb & 1;
    const bool pf = (kb + 1 < nkb);
    float4 kreg[8], vreg[8];
    if (pf) {  // prefetch next K/V tile into registers
#pragma unroll
      for (int u = 0; u < 8; ++u) {
        const int i = tid + u * 128, row = i >> 4, col = (i & 15) * 4;
        const size_t goff = (size_t)((kb + 1) * 64 + row) * ld + col;
        kreg[u] = *(const float4*)(Kg + goff);
        vreg[u] = *(const float4*)(Vg + goff);
      }
    }

    // S = Q * K^T : 4 tiles of 16 key cols, K-dim (d) = 64 -> 2 WMMA each.
    v8f s[4];
#pragma unroll
    for (int j = 0; j < 4; ++j) {
      v8f a = zero8();
      v16bf bk0 = frag_b(&sK[cur][(j * 16) * FSTR], FSTR);
      v16bf bk1 = frag_b(&sK[cur][(j * 16) * FSTR + 32], FSTR);
      a = wmma_bf16(aq0, bk0, a);
      a = wmma_bf16(aq1, bk1, a);
      s[j] = a;
    }

    if (causal && kb == qb) {  // only the diagonal block is partial
#pragma unroll
      for (int j = 0; j < 4; ++j)
#pragma unroll
        for (int r = 0; r < 8; ++r) {
          const int col = kb * 64 + j * 16 + ln;
          const int row = qb * 64 + wave * 16 + r + 8 * lg;
          if (col > row) s[j][r] = -1.0e30f;
        }
    }

    // Online softmax.
#pragma unroll
    for (int r = 0; r < 8; ++r) {
      float vmax = fmaxf(fmaxf(s[0][r], s[1][r]), fmaxf(s[2][r], s[3][r]));
      vmax = fmaxf(vmax, __shfl_xor(vmax, 1));
      vmax = fmaxf(vmax, __shfl_xor(vmax, 2));
      vmax = fmaxf(vmax, __shfl_xor(vmax, 4));
      vmax = fmaxf(vmax, __shfl_xor(vmax, 8));
      const float mnew  = fmaxf(mr[r], vmax);
      const float alpha = __expf(mr[r] - mnew);
      mr[r] = mnew;
      float ps = 0.f;
#pragma unroll
      for (int j = 0; j < 4; ++j) {
        const float p = __expf(s[j][r] - mnew);
        s[j][r] = p;
        ps += p;
      }
      ps += __shfl_xor(ps, 1); ps += __shfl_xor(ps, 2);
      ps += __shfl_xor(ps, 4); ps += __shfl_xor(ps, 8);
      lr[r] = lr[r] * alpha + ps;
      o0[r] *= alpha; o1[r] *= alpha; o2[r] *= alpha; o3[r] *= alpha;
    }

    // Re-layout P (C-layout -> A-frag) through wave-private LDS.  Same-wave
    // DS ops are in-order (DScnt), so no __syncthreads() is required here.
    bf16* pb = &sP[wave * 16 * FSTR];
#pragma unroll
    for (int j = 0; j < 4; ++j)
#pragma unroll
      for (int r = 0; r < 8; ++r)
        pb[(r + 8 * lg) * FSTR + j * 16 + ln] = (bf16)s[j][r];
    const v16bf p0 = frag_a(pb, FSTR);        // keys 0..31
    const v16bf p1 = frag_a(pb + 32, FSTR);   // keys 32..63

    // O += P * V : B-frags from transposed V (row = d, col = key).
    v16bf bv;
    bv = frag_b(&sVt[cur][0 * 16 * FSTR], FSTR);       o0 = wmma_bf16(p0, bv, o0);
    bv = frag_b(&sVt[cur][0 * 16 * FSTR + 32], FSTR);  o0 = wmma_bf16(p1, bv, o0);
    bv = frag_b(&sVt[cur][1 * 16 * FSTR], FSTR);       o1 = wmma_bf16(p0, bv, o1);
    bv = frag_b(&sVt[cur][1 * 16 * FSTR + 32], FSTR);  o1 = wmma_bf16(p1, bv, o1);
    bv = frag_b(&sVt[cur][2 * 16 * FSTR], FSTR);       o2 = wmma_bf16(p0, bv, o2);
    bv = frag_b(&sVt[cur][2 * 16 * FSTR + 32], FSTR);  o2 = wmma_bf16(p1, bv, o2);
    bv = frag_b(&sVt[cur][3 * 16 * FSTR], FSTR);       o3 = wmma_bf16(p0, bv, o3);
    bv = frag_b(&sVt[cur][3 * 16 * FSTR + 32], FSTR);  o3 = wmma_bf16(p1, bv, o3);

    if (pf) {  // store prefetched tile into the other buffer
#pragma unroll
      for (int u = 0; u < 8; ++u) {
        const int i = tid + u * 128, row = i >> 4, col = (i & 15) * 4;
        cvt_store4(&sK[cur ^ 1][row * FSTR + col], kreg[u]);
        sVt[cur ^ 1][(col + 0) * FSTR + row] = (bf16)vreg[u].x;
        sVt[cur ^ 1][(col + 1) * FSTR + row] = (bf16)vreg[u].y;
        sVt[cur ^ 1][(col + 2) * FSTR + row] = (bf16)vreg[u].z;
        sVt[cur ^ 1][(col + 3) * FSTR + row] = (bf16)vreg[u].w;
      }
    }
    __syncthreads();
  }

  // Normalize and write O (same (B,L,H,64) layout as Q).
#pragma unroll
  for (int r = 0; r < 8; ++r) {
    const float inv = 1.0f / lr[r];
    const int trow = wave * 16 + r + 8 * lg;
    float* op = Og + (size_t)trow * ld;
    op[0 * 16 + ln] = o0[r] * inv;
    op[1 * 16 + ln] = o1[r] * inv;
    op[2 * 16 + ln] = o2[r] * inv;
    op[3 * 16 + ln] = o3[r] * inv;
  }
}

// ---------------------------------------------------------------------------
// RoPE (in-place, reference semantics):
//   s = dis[0:32], c = dis[32:64]; x0 = x[0::2], x1 = x[1::2]
//   out = [x0*c - x1*s, x1*c + x0*s]
// One thread per (row, head) block of 64 contiguous floats.
// ---------------------------------------------------------------------------
__global__ __launch_bounds__(256) void rope_kernel(float* __restrict__ x,
                                                   const float* __restrict__ dis,
                                                   int nblk) {
  const int idx = blockIdx.x * blockDim.x + threadIdx.x;
  if (idx >= nblk) return;
  float* xp = x + (size_t)idx * 64;
  const float* dp = dis + (size_t)idx * 64;
  float out[64];
#pragma unroll
  for (int i = 0; i < 32; ++i) {
    const float s = dp[i], c = dp[32 + i];
    const float x0 = xp[2 * i], x1 = xp[2 * i + 1];
    out[i]      = x0 * c - x1 * s;
    out[32 + i] = x1 * c + x0 * s;
  }
#pragma unroll
  for (int i = 0; i < 64; ++i) xp[i] = out[i];
}

// ---------------------------------------------------------------------------
// deep_norm: out = LayerNorm(x*2 + post) * g + b    (H = 512, eps = 1e-5)
// One 256-thread block per row; 2 elements per thread; LDS tree reduction.
// ---------------------------------------------------------------------------
__global__ __launch_bounds__(256) void deepnorm_kernel(
    const float* __restrict__ x, const float* __restrict__ post,
    const float* __restrict__ g, const float* __restrict__ bsh,
    float* __restrict__ out) {
  const int H = 512;
  const int row = blockIdx.x;
  const int t = threadIdx.x;
  __shared__ float rs[256];
  __shared__ float rq[256];
  const float* xp = x + (size_t)row * H;
  const float* pp = post + (size_t)row * H;
  const float v0 = xp[t] * 2.0f + pp[t];
  const float v1 = xp[t + 256] * 2.0f + pp[t + 256];
  rs[t] = v0 + v1;
  rq[t] = v0 * v0 + v1 * v1;
  __syncthreads();
  for (int off = 128; off > 0; off >>= 1) {
    if (t < off) { rs[t] += rs[t + off]; rq[t] += rq[t + off]; }
    __syncthreads();
  }
  const float mu   = rs[0] * (1.0f / H);
  const float var  = rq[0] * (1.0f / H) - mu * mu;
  const float rinv = rsqrtf(var + 1e-5f);
  float* op = out + (size_t)row * H;
  op[t]       = (v0 - mu) * rinv * g[t] + bsh[t];
  op[t + 256] = (v1 - mu) * rinv * g[t + 256] + bsh[t + 256];
}

__global__ void copy_scalar_kernel(const float* __restrict__ a,
                                   float* __restrict__ o) { o[0] = a[0]; }

// ---------------------------------------------------------------------------
// Orchestration
// ---------------------------------------------------------------------------
extern "C" void kernel_launch(void* const* d_in, const int* in_sizes, int n_in,
                              void* d_out, int out_size, void* d_ws, size_t ws_size,
                              hipStream_t stream) {
  (void)in_sizes; (void)n_in; (void)out_size; (void)ws_size;

  const float* tgt  = (const float*)d_in[0];
  const float* mem  = (const float*)d_in[1];
  const float* gm   = (const float*)d_in[2];   // glycan_mass       (self K rope)
  const float* gqm  = (const float*)d_in[3];   // glycan_query_mass (self Q rope)
  const float* gca  = (const float*)d_in[4];   // cross Q rope
  const float* nme  = (const float*)d_in[5];   // cross K rope
  const float* Ain  = (const float*)d_in[10];
  const float* wq1  = (const float*)d_in[11]; const float* bq1 = (const float*)d_in[12];
  const float* wk1  = (const float*)d_in[13]; const float* bk1 = (const float*)d_in[14];
  const float* wv1  = (const float*)d_in[15]; const float* bv1 = (const float*)d_in[16];
  const float* wo1  = (const float*)d_in[17]; const float* bo1 = (const float*)d_in[18];
  const float* l1g  = (const float*)d_in[19]; const float* l1b = (const float*)d_in[20];
  const float* wq2  = (const float*)d_in[21]; const float* bq2 = (const float*)d_in[22];
  const float* wk2  = (const float*)d_in[23]; const float* bk2 = (const float*)d_in[24];
  const float* wv2  = (const float*)d_in[25]; const float* bv2 = (const float*)d_in[26];
  const float* wo2  = (const float*)d_in[27]; const float* bo2 = (const float*)d_in[28];
  const float* l2g  = (const float*)d_in[29]; const float* l2b = (const float*)d_in[30];
  const float* wff1 = (const float*)d_in[31]; const float* wff2 = (const float*)d_in[32];
  const float* l3g  = (const float*)d_in[33]; const float* l3b = (const float*)d_in[34];

  float* out = (float*)d_out;

  const int MQ = 4 * 512;   // B*LT = 2048
  const int MK = 4 * 1024;  // B*LM = 4096

  char* ws = (char*)d_ws;
  size_t off = 0;
  auto alloc = [&](size_t nfloat) {
    float* p = (float*)(ws + off);
    off += nfloat * sizeof(float);
    return p;
  };
  float* q1  = alloc((size_t)MQ * 512);
  float* k1  = alloc((size_t)MQ * 512);
  float* v1  = alloc((size_t)MQ * 512);
  float* ao1 = alloc((size_t)MQ * 512);
  float* po1 = alloc((size_t)MQ * 512);
  float* x1  = alloc((size_t)MQ * 512);
  float* q2  = alloc((size_t)MQ * 1024);
  float* k2  = alloc((size_t)MK * 1024);
  float* v2  = alloc((size_t)MK * 1024);
  float* ao2 = alloc((size_t)MQ * 1024);
  float* po2 = alloc((size_t)MQ * 512);
  float* x2  = alloc((size_t)MQ * 512);
  float* ffh = alloc((size_t)MQ * 2048);
  float* ffo = alloc((size_t)MQ * 512);

  auto gemm = [&](const float* A, const float* W, const float* bias, float* C,
                  int M, int N, int K, int relu) {
    dim3 grid(N / GBN, M / GBM);
    gemm_nt_kernel<<<grid, 256, 0, stream>>>(A, W, bias, C, M, N, K, relu);
  };

  // ---- Self attention ----
  gemm(tgt, wq1, bq1, q1, MQ, 512, 512, 0);
  gemm(tgt, wk1, bk1, k1, MQ, 512, 512, 0);
  gemm(tgt, wv1, bv1, v1, MQ, 512, 512, 0);
  rope_kernel<<<(MQ * 8 + 255) / 256, 256, 0, stream>>>(q1, gqm, MQ * 8);
  rope_kernel<<<(MQ * 8 + 255) / 256, 256, 0, stream>>>(k1, gm, MQ * 8);
  flash_attn_kernel<<<dim3(512 / 64, 4 * 8), 128, 0, stream>>>(
      q1, k1, v1, ao1, 8, 512, 512, /*causal=*/1, 0.125f);
  gemm(ao1, wo1, bo1, po1, MQ, 512, 512, 0);
  deepnorm_kernel<<<MQ, 256, 0, stream>>>(tgt, po1, l1g, l1b, x1);

  // ---- Cross attention ----
  gemm(x1,  wq2, bq2, q2, MQ, 1024, 512, 0);
  gemm(mem, wk2, bk2, k2, MK, 1024, 512, 0);
  gemm(mem, wv2, bv2, v2, MK, 1024, 512, 0);
  rope_kernel<<<(MQ * 16 + 255) / 256, 256, 0, stream>>>(q2, gca, MQ * 16);
  rope_kernel<<<(MK * 16 + 255) / 256, 256, 0, stream>>>(k2, nme, MK * 16);
  flash_attn_kernel<<<dim3(512 / 64, 4 * 16), 128, 0, stream>>>(
      q2, k2, v2, ao2, 16, 512, 1024, /*causal=*/0, 0.125f);
  gemm(ao2, wo2, bo2, po2, MQ, 512, 1024, 0);
  deepnorm_kernel<<<MQ, 256, 0, stream>>>(x1, po2, l2g, l2b, x2);

  // ---- FFN ----
  gemm(x2,  wff1, nullptr, ffh, MQ, 2048, 512, /*relu=*/1);
  gemm(ffh, wff2, nullptr, ffo, MQ, 512, 2048, 0);
  deepnorm_kernel<<<MQ, 256, 0, stream>>>(x2, ffo, l3g, l3b, out);

  // ---- Passthrough scalar A ----
  copy_scalar_kernel<<<1, 1, 0, stream>>>(Ain, out + (size_t)MQ * 512);
}
